// xLSTMBlock_37374805409911
// MI455X (gfx1250) — compile-verified, matrix operand order checked by reading, not run on hardware
//
#include <hip/hip_runtime.h>

// ---------------- problem constants ----------------
#define B_   4
#define S_   2048
#define D_   768
#define I_   1536
#define H_   8
#define HD_  64
#define M_   (B_ * S_)        // 8192 tokens
#define QKV_ (H_ * HD_)       // 512
#define EPS_ 1e-5f

typedef __bf16 bf16_t;
typedef __attribute__((ext_vector_type(8)))  __bf16 v8bf;
typedef __attribute__((ext_vector_type(16))) __bf16 v16bf;
typedef __attribute__((ext_vector_type(8)))  float  v8f;
typedef unsigned int v4u __attribute__((ext_vector_type(4)));
typedef int          v4i __attribute__((ext_vector_type(4)));
typedef int          v8i __attribute__((ext_vector_type(8)));

#if __has_builtin(__builtin_amdgcn_tensor_load_to_lds)
#define HAVE_TDM 1
#else
#define HAVE_TDM 0
#endif

// ---------------------------------------------------------------------------
// Weight convert + transpose: W (K x N, f32 row-major) -> Wt (N x K, bf16)
// so that WMMA B-fragments / TDM rows are contiguous along K.
// ---------------------------------------------------------------------------
__global__ void wcvt_kernel(const float* __restrict__ W, bf16_t* __restrict__ Wt,
                            int K, int N) {
  size_t total = (size_t)K * (size_t)N;
  for (size_t idx = (size_t)blockIdx.x * blockDim.x + threadIdx.x; idx < total;
       idx += (size_t)gridDim.x * blockDim.x) {
    size_t k = idx / (size_t)N;
    size_t n = idx % (size_t)N;
    Wt[n * (size_t)K + k] = (bf16_t)W[idx];
  }
}

// ---------------------------------------------------------------------------
// LayerNorm over D=768, output bf16. One 256-thread block per token.
// ---------------------------------------------------------------------------
__global__ void ln_kernel(const float* __restrict__ x, const float* __restrict__ g,
                          const float* __restrict__ b, bf16_t* __restrict__ h) {
  const int token = blockIdx.x;
  const int t = threadIdx.x;
  __shared__ float red[256];
  const float* xr = x + (size_t)token * D_;

  float s = 0.f;
  for (int i = t; i < D_; i += 256) s += xr[i];
  red[t] = s; __syncthreads();
  for (int off = 128; off > 0; off >>= 1) {
    if (t < off) red[t] += red[t + off];
    __syncthreads();
  }
  const float mu = red[0] * (1.0f / D_);
  __syncthreads();

  float s2 = 0.f;
  for (int i = t; i < D_; i += 256) { float d = xr[i] - mu; s2 += d * d; }
  red[t] = s2; __syncthreads();
  for (int off = 128; off > 0; off >>= 1) {
    if (t < off) red[t] += red[t + off];
    __syncthreads();
  }
  const float rstd = rsqrtf(red[0] * (1.0f / D_) + EPS_);

  for (int i = t; i < D_; i += 256)
    h[(size_t)token * D_ + i] = (bf16_t)((xr[i] - mu) * rstd * g[i] + b[i]);
}

// ---------------------------------------------------------------------------
// WMMA GEMM: out(M x N, f32 acc) = A(M x K, bf16 row-major) @ Wt(N x K, bf16).
// 256-thread block (8 waves) computes a 128(M) x 64(N) macro-tile.
// The shared B k-slice (64 cols x 32 K = 4KB) is staged into LDS by the
// Tensor Data Mover (double-buffered, overlapped with WMMA compute); all 8
// waves consume it via ds_load_b128, cutting B traffic 8x. A-rows stream
// directly from global (per-wave-unique) with global_prefetch_b8.
// Fragment layouts per CDNA5 ISA §7.12.2.
// ---------------------------------------------------------------------------
__global__ void __launch_bounds__(256)
wmma_gemm_bf16(const bf16_t* __restrict__ A, const bf16_t* __restrict__ Bt,
               const float* __restrict__ bias, const float* __restrict__ resid,
               float* __restrict__ outF, bf16_t* __restrict__ outH,
               int Kdim, int Ndim, float scale, int do_sigmoid) {
  __shared__ bf16_t Bs[2][64 * 32];   // double-buffered B slice, 2 x 4KB

  const int tid  = threadIdx.x;
  const int wave = tid >> 5;          // 0..7 -> M sub-tile
  const int lane = tid & 31;
  const int half = lane >> 4;         // K sub-range select per ISA layout
  const int r    = lane & 15;         // row (A) / col (B,C)
  const int m0   = blockIdx.y * 128 + wave * 16;
  const int n0   = blockIdx.x << 6;
  const int nch  = Kdim >> 5;         // number of K=32 chunks

  v8f acc[4] = {v8f{0.f}, v8f{0.f}, v8f{0.f}, v8f{0.f}};
  const bf16_t* Arow = A + (size_t)(m0 + r) * Kdim;

#if HAVE_TDM
  // ---- TDM descriptor: 2D tile, 64 rows (cols of W) x 32 elems (K), 2B ----
  // group1 is chunk-invariant.
  v8i g1;
  g1[0] = 1 << 16;                          // data_size = 1 -> 2 bytes
  g1[1] = 32 << 16;                         // tensor_dim0[15:0]   = 32
  g1[2] = 64 << 16;                         // tensor_dim1[15:0]   = 64
  g1[3] = 32 << 16;                         // tile_dim0           = 32
  g1[4] = 64;                               // tile_dim1           = 64
  g1[5] = Kdim;                             // tensor_dim0_stride  = K elems
  g1[6] = 0;
  g1[7] = 0;
  const v4i gz = {0, 0, 0, 0};              // groups 2/3 unused (2D tensor)

  auto tdm_issue = [&](int t) {
    unsigned long long ga =
        (unsigned long long)(size_t)(Bt + (size_t)n0 * Kdim + (size_t)t * 32);
    unsigned int la = (unsigned int)(size_t)&Bs[t & 1][0];  // LDS byte offset
    v4u g0;
    g0[0] = 1u;                                             // count=1, user D#
    g0[1] = la;                                             // lds_addr
    g0[2] = (unsigned int)ga;                               // global_addr[31:0]
    g0[3] = (unsigned int)((ga >> 32) & 0x01FFFFFFu) | (2u << 30); // [56:32]|type=2
#if defined(__clang_major__) && (__clang_major__ >= 23)
    __builtin_amdgcn_tensor_load_to_lds(g0, g1, gz, gz, g1, 0);   // 6-arg form
#else
    __builtin_amdgcn_tensor_load_to_lds(g0, g1, gz, gz, 0);       // 5-arg form
#endif
  };

  if (wave == 0) tdm_issue(0);
#endif

  for (int t = 0; t < nch; ++t) {
    const int k0 = t << 5;
#if HAVE_TDM
    if (wave == 0) {
      if (t + 1 < nch) {
        tdm_issue(t + 1);                          // prefetch next slice
        __builtin_amdgcn_s_wait_tensorcnt(1);      // in-order TDM: slice t done
      } else {
        __builtin_amdgcn_s_wait_tensorcnt(0);
      }
    }
    __syncthreads();                               // publish LDS to all waves
#else
    // Cooperative fallback copy: 256 threads x 16B = 4KB slice.
    {
      const int row = tid >> 2, seg = tid & 3;
      *(v8bf*)&Bs[t & 1][row * 32 + seg * 8] =
          *(const v8bf*)(Bt + (size_t)(n0 + row) * Kdim + k0 + seg * 8);
    }
    __syncthreads();
#endif

    // A 16x32 bf16 fragment: lane r = row, half picks K {0..7,16..23}/{8..15,24..31}
    v8bf alo = *(const v8bf*)(Arow + k0 + 8 * half);
    v8bf ahi = *(const v8bf*)(Arow + k0 + 16 + 8 * half);
    v16bf afrag = __builtin_shufflevector(alo, ahi,
        0,1,2,3,4,5,6,7,8,9,10,11,12,13,14,15);
    __builtin_prefetch(Arow + k0 + 128, 0, 1);     // global_prefetch_b8

#pragma unroll
    for (int j = 0; j < 4; ++j) {
      const bf16_t* brow = &Bs[t & 1][(16 * j + r) * 32];
      v8bf blo = *(const v8bf*)(brow + 8 * half);
      v8bf bhi = *(const v8bf*)(brow + 16 + 8 * half);
      v16bf bfrag = __builtin_shufflevector(blo, bhi,
          0,1,2,3,4,5,6,7,8,9,10,11,12,13,14,15);
      acc[j] = __builtin_amdgcn_wmma_f32_16x16x32_bf16(
          /*neg_a=*/false, afrag, /*neg_b=*/false, bfrag,
          /*c_mod=*/(short)0, acc[j], /*reuse_a=*/false, /*reuse_b=*/false);
    }
    __syncthreads();   // all waves done with buf[t&1] before TDM overwrites it
  }

#pragma unroll
  for (int j = 0; j < 4; ++j) {
    const int n = n0 + (j << 4) + r;
    const float bv = bias ? bias[n] : 0.0f;
#pragma unroll
    for (int rr = 0; rr < 8; ++rr) {
      const int mm = m0 + (half << 3) + rr;        // C layout: VGPR rr -> M=8*half+rr
      float val = acc[j][rr] * scale + bv;
      if (do_sigmoid) val = 1.0f / (1.0f + __expf(-val));
      const size_t off = (size_t)mm * Ndim + n;
      if (resid) val += resid[off];
      if (outF) outF[off] = val;
      if (outH) outH[off] = (bf16_t)val;
    }
  }
}

// ---------------------------------------------------------------------------
// Gate pre-activations: i_raw/f_raw (M x 8) = u @ Wi/Wf + bias. Tiny N; scalar.
// ---------------------------------------------------------------------------
__global__ void gates_kernel(const bf16_t* __restrict__ u,
                             const float* __restrict__ Wi, const float* __restrict__ bi,
                             const float* __restrict__ Wf, const float* __restrict__ bfv,
                             float* __restrict__ ir, float* __restrict__ fr) {
  const int token = blockIdx.x;
  const int t = threadIdx.x;    // 0..63
  __shared__ float red[64];
  const bf16_t* ur = u + (size_t)token * I_;
  for (int h = 0; h < H_; ++h) {
    float ai = 0.f, af = 0.f;
    for (int k = t; k < I_; k += 64) {
      float uk = (float)ur[k];
      ai += uk * Wi[(size_t)k * H_ + h];
      af += uk * Wf[(size_t)k * H_ + h];
    }
    red[t] = ai; __syncthreads();
    if (t == 0) {
      float s = 0.f;
      for (int j2 = 0; j2 < 64; ++j2) s += red[j2];
      ir[(size_t)token * H_ + h] = s + bi[h];
    }
    __syncthreads();
    red[t] = af; __syncthreads();
    if (t == 0) {
      float s = 0.f;
      for (int j2 = 0; j2 < 64; ++j2) s += red[j2];
      fr[(size_t)token * H_ + h] = s + bfv[h];
    }
    __syncthreads();
  }
}

// ---------------------------------------------------------------------------
// Sequential mLSTM scan. One block per (b,h); 64 threads (2 waves).
// Thread t owns C[t][0..63] in registers (v index) and n[t] (k index).
// ---------------------------------------------------------------------------
__global__ void __launch_bounds__(64)
scan_kernel(const float* __restrict__ q, const float* __restrict__ k,
            const float* __restrict__ v, const float* __restrict__ ir,
            const float* __restrict__ fr, float* __restrict__ ht) {
  const int bh = blockIdx.x;            // 0..B*H-1
  const int b  = bh / H_;
  const int h  = bh % H_;
  const int t  = threadIdx.x;           // 0..63

  float Crow[HD_];
#pragma unroll
  for (int i = 0; i < HD_; ++i) Crow[i] = 0.f;
  float n_t = 0.f;

  __shared__ float qs[HD_], ks[HD_], vs[HD_], red[HD_];
  __shared__ float m_s, den_s;
  if (t == 0) m_s = 0.f;
  __syncthreads();

  for (int s = 0; s < S_; ++s) {
    const size_t tok  = (size_t)b * S_ + s;
    const size_t base = (tok * H_ + h) * HD_ + t;
    qs[t] = q[base]; ks[t] = k[base]; vs[t] = v[base];
    __syncthreads();

    const float it = ir[tok * H_ + h];
    const float ft = fr[tok * H_ + h];
    const float m_old = m_s;
    const float m_new = fmaxf(ft + m_old, it);
    const float ig = __expf(it - m_new);
    const float fg = __expf(ft + m_old - m_new);

    n_t = fg * n_t + ig * ks[t];
    red[t] = n_t * qs[t];
    __syncthreads();
    if (t == 0) {
      float d = 0.f;
      for (int i = 0; i < HD_; ++i) d += red[i];
      den_s = fmaxf(fabsf(d), 1.0f);
      m_s = m_new;
    }
    __syncthreads();

    const float vt = vs[t];
    float num = 0.f;
#pragma unroll
    for (int kk = 0; kk < HD_; ++kk) {
      Crow[kk] = fg * Crow[kk] + ig * vt * ks[kk];
      num += Crow[kk] * qs[kk];
    }
    ht[base] = num / den_s;
    __syncthreads();
  }
}

// ---------------------------------------------------------------------------
// Per-head GroupNorm (scale-only) fused with output gate: a = sigmoid_o * hn.
// ---------------------------------------------------------------------------
__global__ void gnorm_kernel(const float* __restrict__ ht, const float* __restrict__ o,
                             const float* __restrict__ mh_g, bf16_t* __restrict__ a) {
  const int g = blockIdx.x;             // token*H + h
  const int h = g % H_;
  const int t = threadIdx.x;            // 0..63
  const size_t base = (size_t)g * HD_ + t;
  __shared__ float red[HD_];
  __shared__ float mu_s, var_s;

  const float xv = ht[base];
  red[t] = xv; __syncthreads();
  if (t == 0) {
    float s = 0.f;
    for (int i = 0; i < HD_; ++i) s += red[i];
    mu_s = s * (1.0f / HD_);
  }
  __syncthreads();
  const float d = xv - mu_s;
  red[t] = d * d; __syncthreads();
  if (t == 0) {
    float s = 0.f;
    for (int i = 0; i < HD_; ++i) s += red[i];
    var_s = s * (1.0f / HD_);
  }
  __syncthreads();
  const float hn = d * rsqrtf(var_s + EPS_) * mh_g[h * HD_ + t];
  a[base] = (bf16_t)(o[base] * hn);
}

// ---------------------------------------------------------------------------
// Host launcher
// ---------------------------------------------------------------------------
extern "C" void kernel_launch(void* const* d_in, const int* in_sizes, int n_in,
                              void* d_out, int out_size, void* d_ws, size_t ws_size,
                              hipStream_t stream) {
  (void)in_sizes; (void)n_in; (void)out_size; (void)ws_size;

  const float* x      = (const float*)d_in[0];
  const float* ln_g   = (const float*)d_in[1];
  const float* ln_b   = (const float*)d_in[2];
  const float* W_up   = (const float*)d_in[3];
  const float* W_down = (const float*)d_in[4];
  const float* Wq     = (const float*)d_in[5];
  const float* Wk     = (const float*)d_in[6];
  const float* Wv     = (const float*)d_in[7];
  const float* Wi     = (const float*)d_in[8];
  const float* bi     = (const float*)d_in[9];
  const float* Wf     = (const float*)d_in[10];
  const float* bfv    = (const float*)d_in[11];
  const float* Wo     = (const float*)d_in[12];
  const float* bo     = (const float*)d_in[13];
  const float* mh_g   = (const float*)d_in[14];
  const float* W_out  = (const float*)d_in[15];
  float* out = (float*)d_out;

  char* p = (char*)d_ws;
  auto take = [&](size_t bytes) -> char* {
    char* q = p;
    p += (bytes + 255) & ~(size_t)255;
    return q;
  };

  bf16_t* hbf     = (bf16_t*)take((size_t)M_ * D_   * sizeof(bf16_t));
  bf16_t* ubf     = (bf16_t*)take((size_t)M_ * I_   * sizeof(bf16_t));
  float*  qf      = (float*) take((size_t)M_ * QKV_ * sizeof(float));
  float*  kf      = (float*) take((size_t)M_ * QKV_ * sizeof(float));
  float*  vf      = (float*) take((size_t)M_ * QKV_ * sizeof(float));
  float*  of      = (float*) take((size_t)M_ * QKV_ * sizeof(float));
  float*  irf     = (float*) take((size_t)M_ * H_   * sizeof(float));
  float*  frf     = (float*) take((size_t)M_ * H_   * sizeof(float));
  float*  htf     = (float*) take((size_t)M_ * QKV_ * sizeof(float));
  bf16_t* abf     = (bf16_t*)take((size_t)M_ * QKV_ * sizeof(bf16_t));
  bf16_t* ybf     = (bf16_t*)take((size_t)M_ * I_   * sizeof(bf16_t));
  bf16_t* Wup_t   = (bf16_t*)take((size_t)D_ * I_   * sizeof(bf16_t));
  bf16_t* Wq_t    = (bf16_t*)take((size_t)I_ * QKV_ * sizeof(bf16_t));
  bf16_t* Wk_t    = (bf16_t*)take((size_t)I_ * QKV_ * sizeof(bf16_t));
  bf16_t* Wv_t    = (bf16_t*)take((size_t)I_ * QKV_ * sizeof(bf16_t));
  bf16_t* Wo_t    = (bf16_t*)take((size_t)I_ * QKV_ * sizeof(bf16_t));
  bf16_t* Wout_t  = (bf16_t*)take((size_t)QKV_ * I_ * sizeof(bf16_t));
  bf16_t* Wdown_t = (bf16_t*)take((size_t)I_ * D_   * sizeof(bf16_t));

  // 1) weight conversion f32 -> bf16, transposed to N x K
  wcvt_kernel<<<512, 256, 0, stream>>>(W_up,   Wup_t,   D_,   I_);
  wcvt_kernel<<<512, 256, 0, stream>>>(Wq,     Wq_t,    I_,   QKV_);
  wcvt_kernel<<<512, 256, 0, stream>>>(Wk,     Wk_t,    I_,   QKV_);
  wcvt_kernel<<<512, 256, 0, stream>>>(Wv,     Wv_t,    I_,   QKV_);
  wcvt_kernel<<<512, 256, 0, stream>>>(Wo,     Wo_t,    I_,   QKV_);
  wcvt_kernel<<<512, 256, 0, stream>>>(W_out,  Wout_t,  QKV_, I_);
  wcvt_kernel<<<512, 256, 0, stream>>>(W_down, Wdown_t, I_,   D_);

  // 2) LayerNorm -> h (bf16)
  ln_kernel<<<M_, 256, 0, stream>>>(x, ln_g, ln_b, hbf);

  // 3) up projection: u = h @ W_up (bf16 out for downstream WMMA)
  wmma_gemm_bf16<<<dim3(I_ / 64, M_ / 128), 256, 0, stream>>>(
      hbf, Wup_t, nullptr, nullptr, nullptr, ubf, D_, I_, 1.0f, 0);

  // 4) q/k/v/o projections (f32 out for the scan; k scaled by HD^-0.5; o sigmoid+bias)
  wmma_gemm_bf16<<<dim3(QKV_ / 64, M_ / 128), 256, 0, stream>>>(
      ubf, Wq_t, nullptr, nullptr, qf, nullptr, I_, QKV_, 1.0f, 0);
  wmma_gemm_bf16<<<dim3(QKV_ / 64, M_ / 128), 256, 0, stream>>>(
      ubf, Wk_t, nullptr, nullptr, kf, nullptr, I_, QKV_, 0.125f /* 64^-0.5 */, 0);
  wmma_gemm_bf16<<<dim3(QKV_ / 64, M_ / 128), 256, 0, stream>>>(
      ubf, Wv_t, nullptr, nullptr, vf, nullptr, I_, QKV_, 1.0f, 0);
  wmma_gemm_bf16<<<dim3(QKV_ / 64, M_ / 128), 256, 0, stream>>>(
      ubf, Wo_t, bo, nullptr, of, nullptr, I_, QKV_, 1.0f, 1);

  // 5) scalar gate pre-activations
  gates_kernel<<<M_, 64, 0, stream>>>(ubf, Wi, bi, Wf, bfv, irf, frf);

  // 6) sequential mLSTM scan, one block per (b,h)
  scan_kernel<<<B_ * H_, 64, 0, stream>>>(qf, kf, vf, irf, frf, htf);

  // 7) GroupNorm + output gate -> a (bf16)
  gnorm_kernel<<<M_ * H_, 64, 0, stream>>>(htf, of, mh_g, abf);

  // 8) y = a @ W_out (bf16 out)
  wmma_gemm_bf16<<<dim3(I_ / 64, M_ / 128), 256, 0, stream>>>(
      abf, Wout_t, nullptr, nullptr, nullptr, ybf, QKV_, I_, 1.0f, 0);

  // 9) out = x + y @ W_down (f32 out, fused residual)
  wmma_gemm_bf16<<<dim3(D_ / 64, M_ / 128), 256, 0, stream>>>(
      ybf, Wdown_t, nullptr, x, out, nullptr, I_, D_, 1.0f, 0);
}
